// HyperGNNConv_10376640987275
// MI455X (gfx1250) — matrix operand order repaired and checked by reading, not compiled, not run on hardware
//
#include <hip/hip_runtime.h>

#define N_NODES  50000
#define N_HEDGES 50000
#define N_INC    400000
#define D_IN     128
#define D_HID    128
#define D_OUT    128

typedef __attribute__((ext_vector_type(16))) _Float16 v16h;
typedef __attribute__((ext_vector_type(8)))  float    v8f;

// ---------------------------------------------------------------------------
// 0) zero accumulator region (float4 grid-stride)
// ---------------------------------------------------------------------------
__global__ void hgc_zero4(float4* __restrict__ p, long n4) {
    long i = (long)blockIdx.x * blockDim.x + threadIdx.x;
    long stride = (long)gridDim.x * blockDim.x;
    float4 z = {0.f, 0.f, 0.f, 0.f};
    for (; i < n4; i += stride) p[i] = z;
}

// ---------------------------------------------------------------------------
// 1) pack W [K x 128] f32 (row-major, K x N) into WMMA B-fragment order, f16.
//    One thread per (n-tile, k-step, lane); each writes one contiguous v16h.
//    Fragment layout (ISA 7.12.2, 16-bit B 32x16): lane = khalf*16 + (N%16),
//    VGPR v<4 holds K = kk + 8*khalf + 2v, v>=4 holds K = kk + 16 + 8*khalf + 2(v-4) (+1 in high half).
// ---------------------------------------------------------------------------
__global__ void hgc_pack_w(const float* __restrict__ W, v16h* __restrict__ Wp,
                           int K) {
    const int ksteps = K / 32;
    int t    = blockIdx.x * blockDim.x + threadIdx.x;
    int lane = t & 31;
    int ks   = (t >> 5) % ksteps;
    int nt   = (t >> 5) / ksteps;
    if (nt >= 8) return;                       // 8 n-tiles (N = 128)
    const int l16   = lane & 15;
    const int khalf = lane >> 4;
    const int n     = nt * 16 + l16;
    const int kk    = ks * 32;
    v16h b;
#pragma unroll
    for (int v = 0; v < 8; ++v) {
        int kb = kk + ((v < 4) ? 2 * v : 16 + 2 * (v - 4)) + 8 * khalf;
        b[2 * v]     = (_Float16)W[(size_t)kb       * 128 + n];
        b[2 * v + 1] = (_Float16)W[(size_t)(kb + 1) * 128 + n];
    }
    Wp[((size_t)nt * ksteps + ks) * 32 + lane] = b;
}

// ---------------------------------------------------------------------------
// 2) scatter-add: one wave (32 lanes) per incidence row, float4 per lane.
//    sum[seg[i]][:] += src[gidx[i]][:]  ;  cnt[seg[i]] += 1
// ---------------------------------------------------------------------------
__global__ void hgc_scatter_add(const int* __restrict__ seg,
                                const int* __restrict__ gidx,
                                const float* __restrict__ src,
                                float* __restrict__ sum,
                                float* __restrict__ cnt,
                                int n_inc) {
    int t    = blockIdx.x * blockDim.x + threadIdx.x;
    int inc  = t >> 5;
    int lane = t & 31;
    if (inc >= n_inc) return;
    int s = seg[inc];
    int g = gidx[inc];
    const float4* sp = (const float4*)(src + (size_t)g * D_IN);
    float4 v = sp[lane];
    float* dp = sum + (size_t)s * D_IN + lane * 4;
    atomicAdd(dp + 0, v.x);
    atomicAdd(dp + 1, v.y);
    atomicAdd(dp + 2, v.z);
    atomicAdd(dp + 3, v.w);
    if (lane == 0) atomicAdd(cnt + s, 1.0f);
}

// ---------------------------------------------------------------------------
// 3) edge GEMM: H = ReLU( (edge_sum/cnt) @ W_edge + b_edge )
//    8 waves/block, wave w -> 16x16 tile at (blockIdx.x*16, w*16). K = 128.
//    B comes pre-packed in fragment order: one v16h load per K-step.
// ---------------------------------------------------------------------------
__global__ void hgc_edge_gemm(const float* __restrict__ Asum,
                              const float* __restrict__ cnt,
                              const v16h* __restrict__ Wp,
                              const float* __restrict__ bias,
                              float* __restrict__ H) {
    const int wave  = threadIdx.x >> 5;
    const int lane  = threadIdx.x & 31;
    const int l16   = lane & 15;
    const int khalf = lane >> 4;
    const int m0    = blockIdx.x * 16;
    const int n0    = wave * 16;
    const int row   = m0 + l16;            // A row for this lane (M = lane%16)
    const int n     = n0 + l16;            // C column for this lane (N = lane%16)

    const float scale = 1.0f / fmaxf(cnt[row], 1.0f);
    const float* Arow = Asum + (size_t)row * D_IN;
    const v16h*  Bp   = Wp + (size_t)wave * (D_IN / 32) * 32 + lane;

    v8f c = {};
#pragma unroll
    for (int kk = 0; kk < D_IN; kk += 32) {
        v16h a;
#pragma unroll
        for (int v = 0; v < 8; ++v) {
            // A 16x32 f16 layout: VGPR v<4 -> K=2v(+1), v>=4 -> K=16+2(v-4)(+1),
            // lane-half adds +8 to K.
            int ka = kk + ((v < 4) ? 2 * v : 16 + 2 * (v - 4)) + 8 * khalf;
            a[2 * v]     = (_Float16)(Arow[ka]     * scale);
            a[2 * v + 1] = (_Float16)(Arow[ka + 1] * scale);
        }
        v16h b = Bp[(kk / 32) * 32];
        c = __builtin_amdgcn_wmma_f32_16x16x32_f16(false, a, false, b,
                                                   (short)0, c, false, false);
    }

    const float bv = bias[n];
#pragma unroll
    for (int r = 0; r < 8; ++r) {
        int m = m0 + r + 8 * khalf;        // C/D layout: VGPR r -> M = r + 8*half
        H[(size_t)m * D_HID + n] = fmaxf(c[r] + bv, 0.0f);
    }
}

// ---------------------------------------------------------------------------
// 4) node GEMM: out = concat(x, node_sum/cnt) @ W_node + b_node.  K = 256.
// ---------------------------------------------------------------------------
__global__ void hgc_node_gemm(const float* __restrict__ X,
                              const float* __restrict__ Nsum,
                              const float* __restrict__ cnt,
                              const v16h* __restrict__ Wp,
                              const float* __restrict__ bias,
                              float* __restrict__ out) {
    const int wave  = threadIdx.x >> 5;
    const int lane  = threadIdx.x & 31;
    const int l16   = lane & 15;
    const int khalf = lane >> 4;
    const int m0    = blockIdx.x * 16;
    const int n0    = wave * 16;
    const int row   = m0 + l16;
    const int n     = n0 + l16;
    const int KTOT  = D_IN + D_HID;        // 256

    const float scale = 1.0f / fmaxf(cnt[row], 1.0f);
    const float* Ax = X    + (size_t)row * D_IN;
    const float* An = Nsum + (size_t)row * D_HID;
    const v16h*  Bp = Wp + (size_t)wave * (KTOT / 32) * 32 + lane;

    v8f c = {};
#pragma unroll
    for (int kk = 0; kk < KTOT; kk += 32) {
        v16h a;
#pragma unroll
        for (int v = 0; v < 8; ++v) {
            int ka = kk + ((v < 4) ? 2 * v : 16 + 2 * (v - 4)) + 8 * khalf;
            if (kk < D_IN) {   // compile-time after unroll: first 128 K from x
                a[2 * v]     = (_Float16)Ax[ka];
                a[2 * v + 1] = (_Float16)Ax[ka + 1];
            } else {           // second 128 K from node mean
                a[2 * v]     = (_Float16)(An[ka - D_IN]     * scale);
                a[2 * v + 1] = (_Float16)(An[ka + 1 - D_IN] * scale);
            }
        }
        v16h b = Bp[(kk / 32) * 32];
        c = __builtin_amdgcn_wmma_f32_16x16x32_f16(false, a, false, b,
                                                   (short)0, c, false, false);
    }

    const float bv = bias[n];
#pragma unroll
    for (int r = 0; r < 8; ++r) {
        int m = m0 + r + 8 * khalf;
        out[(size_t)m * D_OUT + n] = c[r] + bv;
    }
}

// ---------------------------------------------------------------------------
// 5) row-wise L2 normalize: one wave per row, float4 per lane, shfl reduction
// ---------------------------------------------------------------------------
__global__ void hgc_l2norm(float* __restrict__ out) {
    int wave = threadIdx.x >> 5;
    int lane = threadIdx.x & 31;
    int row  = blockIdx.x * 8 + wave;          // 6250 blocks * 8 waves = 50000
    float4* p = (float4*)(out + (size_t)row * D_OUT) + lane;
    float4 v = *p;
    float s = v.x * v.x + v.y * v.y + v.z * v.z + v.w * v.w;
#pragma unroll
    for (int off = 16; off; off >>= 1) s += __shfl_xor(s, off, 32);
    float inv = rsqrtf(s);
    v.x *= inv; v.y *= inv; v.z *= inv; v.w *= inv;
    *p = v;
}

// ---------------------------------------------------------------------------
extern "C" void kernel_launch(void* const* d_in, const int* in_sizes, int n_in,
                              void* d_out, int out_size, void* d_ws, size_t ws_size,
                              hipStream_t stream) {
    const float* x      = (const float*)d_in[0];
    const int*   idx    = (const int*)  d_in[1];   // [2, N_INC]: row0 = he, row1 = nd
    const float* W_edge = (const float*)d_in[2];
    const float* b_edge = (const float*)d_in[3];
    const float* W_node = (const float*)d_in[4];
    const float* b_node = (const float*)d_in[5];
    float* out = (float*)d_out;

    const int* he = idx;
    const int* nd = idx + N_INC;

    // workspace layout: accumulators first (contiguous, zeroed every launch),
    // then buffers that are fully overwritten each launch.
    float* edge_sum = (float*)d_ws;                          // 6.40M f32
    float* node_sum = edge_sum + (size_t)N_HEDGES * D_IN;    // 6.40M f32
    float* edge_cnt = node_sum + (size_t)N_NODES * D_HID;    // 50K  f32
    float* node_cnt = edge_cnt + N_HEDGES;                   // 50K  f32
    float* h_edge   = node_cnt + N_NODES;                    // 6.40M f32
    v16h*  wp_edge  = (v16h*)(h_edge + (size_t)N_HEDGES * D_HID); // 128x128 f16
    v16h*  wp_node  = wp_edge + (size_t)(D_IN * D_HID) / 16;      // 256x128 f16

    const long zero_floats = (long)N_HEDGES * D_IN + (long)N_NODES * D_HID
                           + N_HEDGES + N_NODES;             // 12.9M, /4 exact
    hgc_zero4<<<4096, 256, 0, stream>>>((float4*)d_ws, zero_floats / 4);

    // pack weights into WMMA B-fragment layout (f16), recomputed every launch
    hgc_pack_w<<<(8 * (D_IN / 32) * 32 + 255) / 256, 256, 0, stream>>>(
        W_edge, wp_edge, D_IN);
    hgc_pack_w<<<(8 * ((D_IN + D_HID) / 32) * 32 + 255) / 256, 256, 0, stream>>>(
        W_node, wp_node, D_IN + D_HID);

    // 1) scatter node feats into hyperedges:  edge_sum[he] += x[nd]
    hgc_scatter_add<<<(N_INC * 32) / 256, 256, 0, stream>>>(
        he, nd, x, edge_sum, edge_cnt, N_INC);

    // 2) edge MLP (mean + linear + ReLU), WMMA
    hgc_edge_gemm<<<N_HEDGES / 16, 256, 0, stream>>>(
        edge_sum, edge_cnt, wp_edge, b_edge, h_edge);

    // 3) scatter hyperedge reps back to nodes:  node_sum[nd] += h_edge[he]
    hgc_scatter_add<<<(N_INC * 32) / 256, 256, 0, stream>>>(
        nd, he, h_edge, node_sum, node_cnt, N_INC);

    // 4) node linear on concat(x, node_mean), WMMA
    hgc_node_gemm<<<N_NODES / 16, 256, 0, stream>>>(
        x, node_sum, node_cnt, wp_node, b_node, out);

    // 5) L2 normalize rows in place
    hgc_l2norm<<<N_NODES / 8, 256, 0, stream>>>(out);
}
